// QuantizedLinear_68393059222043
// MI455X (gfx1250) — compile-verified, matrix operand order checked by reading, not run on hardware
//
#include <hip/hip_runtime.h>

#define OUT_DIM 8192
#define IN_DIM  8192
#define QGROUP  128
#define NGROUPS (IN_DIM / QGROUP)   // 64
#define BATCH   8
#define ROWS_PER_WG 128
#define KSPLIT  4
#define KSPAN   (IN_DIM / KSPLIT)   // 2048
#define NCHUNK  (KSPAN / QGROUP)    // 16
#define QROW_LDS 68                 // 64 data bytes + 4 pad (bank-conflict-free)
#define XROW_LDS 136                // 128 halves + 8 pad

typedef _Float16 v16h  __attribute__((ext_vector_type(16)));
typedef __fp16   fp16x2 __attribute__((ext_vector_type(2)));  // cvt_pkrtz return type
typedef float    v8f   __attribute__((ext_vector_type(8)));
typedef float    f32x4 __attribute__((ext_vector_type(4)));
typedef unsigned int u32x4 __attribute__((ext_vector_type(4)));
typedef unsigned int u32x2 __attribute__((ext_vector_type(2)));
typedef int      i32x8 __attribute__((ext_vector_type(8)));
typedef int      i32x4 __attribute__((ext_vector_type(4)));

#if defined(__has_builtin)
#if __has_builtin(__builtin_amdgcn_tensor_load_to_lds) && __has_builtin(__builtin_amdgcn_s_wait_tensorcnt)
#define USE_TDM 1
#else
#define USE_TDM 0
#endif
#else
#define USE_TDM 0
#endif

union HalfPack { fp16x2 h; unsigned int u; };
union FragA { v16h v; u32x4 q[2]; };
union FragB { v16h v; unsigned int u[8]; };

// Dequant 8 nibbles of one dword -> 8 f16 (4 dwords), low nibble = lower K.
// w = (q - z) * s computed as fma(2^23 + q, s, -(2^23 + z)*s); exact for q in [0,15].
__device__ __forceinline__ void dq8(unsigned int p, float s, float bias, unsigned int* d) {
#pragma unroll
  for (int i = 0; i < 4; ++i) {
    const unsigned int n0 = (p >> (8 * i)) & 0xFu;
    const unsigned int n1 = (p >> (8 * i + 4)) & 0xFu;
    const float f0 = __uint_as_float(0x4B000000u | n0);
    const float f1 = __uint_as_float(0x4B000000u | n1);
    HalfPack hp;
    hp.h = __builtin_amdgcn_cvt_pkrtz(fmaf(f0, s, bias), fmaf(f1, s, bias));
    d[i] = hp.u;
  }
}

#if USE_TDM
// Issue one TDM 2D tile load: 128 rows x 64 bytes (data_size = 8B units),
// global row stride = 4096 B (512 units), LDS pad 4B per 64B -> 68B LDS rows.
__device__ __forceinline__ void tdm_issue(unsigned long long ga, unsigned int ldsoff) {
  u32x4 g0;
  g0.x = 1u;                                   // count=1, user descriptor
  g0.y = ldsoff;                               // lds_addr (bytes)
  g0.z = (unsigned int)ga;                     // global_addr[31:0]
  g0.w = (unsigned int)(ga >> 32) | (2u << 30);// global_addr[56:32] | type=2
  i32x8 g1;
  g1[0] = (int)((3u << 16) |                   // data_size = 8B
                (1u << 20) |                   // pad_enable
                (3u << 22) |                   // pad_interval: 16 DWORDs = 64B
                (0u << 25));                   // pad_amount: 1 DWORD = 4B
  g1[1] = (int)(512u << 16);                   // tensor_dim0 = 512 (8B units)
  g1[2] = (int)(8192u << 16);                  // tensor_dim1 = 8192 rows
  g1[3] = (int)(8u << 16);                     // tile_dim0 = 8 (64B)
  g1[4] = 128;                                 // tile_dim1 = 128 rows
  g1[5] = 512;                                 // tensor_dim0_stride = 512 units
  g1[6] = 0;
  g1[7] = 0;
  i32x4 g2 = {0, 0, 0, 0};
  i32x4 g3 = {0, 0, 0, 0};
#if defined(__clang_major__) && (__clang_major__ >= 23)
  i32x8 g4 = {0, 0, 0, 0, 0, 0, 0, 0};
  __builtin_amdgcn_tensor_load_to_lds(g0, g1, g2, g3, g4, 0);
#else
  __builtin_amdgcn_tensor_load_to_lds(g0, g1, g2, g3, 0);
#endif
}
#else
__device__ __forceinline__ void fb_load(const unsigned char* __restrict__ g,
                                        unsigned char* l, int tid) {
#pragma unroll
  for (int it = 0; it < 2; ++it) {
    const int id = tid + it * 256;             // 512 x 16B = 8KB tile
    const int row = id >> 2, seg = id & 3;
    const u32x4 v = *(const u32x4*)(g + (size_t)row * (IN_DIM / 2) + seg * 16);
    unsigned int* d = (unsigned int*)(l + row * QROW_LDS + seg * 16);
    d[0] = v.x; d[1] = v.y; d[2] = v.z; d[3] = v.w;
  }
}
#endif

// Stage x[0..7, kbase..kbase+127] as f16 into padded LDS rows.
__device__ __forceinline__ void stage_x(const float* __restrict__ x, _Float16* dst,
                                        int kbase, int tid) {
  const int m = tid >> 5;                      // 0..7 (batch row)
  const int i = tid & 31;                      // 4 floats each
  const f32x4 v = *(const f32x4*)(x + (size_t)m * IN_DIM + kbase + i * 4);
  HalfPack p0, p1;
  p0.h = __builtin_amdgcn_cvt_pkrtz(v.x, v.y);
  p1.h = __builtin_amdgcn_cvt_pkrtz(v.z, v.w);
  u32x2 st; st.x = p0.u; st.y = p1.u;
  *(u32x2*)(dst + m * XROW_LDS + i * 4) = st;
}

__global__ void qlinear_zero_kernel(float* __restrict__ out, int n) {
  const int i = blockIdx.x * blockDim.x + threadIdx.x;
  if (i < n) out[i] = 0.0f;
}

__global__ __launch_bounds__(256)
void qlinear_w4a16_wmma_kernel(const float* __restrict__ x,
                               const unsigned char* __restrict__ qweight,
                               const float* __restrict__ scale,
                               const float* __restrict__ zero,
                               float* __restrict__ out) {
  __shared__ __align__(16) unsigned char qtile[2][ROWS_PER_WG * QROW_LDS]; // 2 x 8704 B
  __shared__ __align__(16) _Float16 xh[2][16 * XROW_LDS];                  // 2 x 4352 B

  const int tid  = threadIdx.x;
  const int lane = tid & 31;
  const int wid  = tid >> 5;          // 8 waves
  const int ln16 = lane & 15;
  const int hl   = lane >> 4;         // half-wave: selects K sub-range of fragment
  const int obase  = blockIdx.x * ROWS_PER_WG;
  const int kbase0 = blockIdx.y * KSPAN;
  const int gbase  = blockIdx.y * NCHUNK;

  // Zero M-padding rows (m = 8..15) of both x buffers once.
  {
    unsigned int* z0 = (unsigned int*)&xh[0][8 * XROW_LDS];
    unsigned int* z1 = (unsigned int*)&xh[1][8 * XROW_LDS];
    for (int t = tid; t < (8 * XROW_LDS) / 2; t += 256) { z0[t] = 0u; z1[t] = 0u; }
  }

  const int r = wid * 16 + ln16;      // row within this WG's 128-row weight tile
  const int orow = obase + r;         // global output column (W row)

  v8f acc = {0.f, 0.f, 0.f, 0.f, 0.f, 0.f, 0.f, 0.f};

  // ---- prologue: chunk 0 -> buffer 0
  {
    const unsigned char* gsrc =
        qweight + (size_t)obase * (IN_DIM / 2) + (kbase0 >> 1);
#if USE_TDM
    if (wid == 0)
      tdm_issue((unsigned long long)gsrc, (unsigned int)(unsigned long long)&qtile[0][0]);
#else
    fb_load(gsrc, &qtile[0][0], tid);
#endif
    stage_x(x, &xh[0][0], kbase0, tid);
  }

  for (int c = 0; c < NCHUNK; ++c) {
    // Prefetch chunk c+1 into the other buffer (safe: last read ended at the
    // trailing barrier of iteration c-1).
    if (c + 1 < NCHUNK) {
      const int nb = (c + 1) & 1;
      const unsigned char* gsrc =
          qweight + (size_t)obase * (IN_DIM / 2) + ((kbase0 + (c + 1) * QGROUP) >> 1);
#if USE_TDM
      if (wid == 0)
        tdm_issue((unsigned long long)gsrc, (unsigned int)(unsigned long long)&qtile[nb][0]);
#else
      fb_load(gsrc, &qtile[nb][0], tid);
#endif
      stage_x(x, &xh[nb][0], kbase0 + (c + 1) * QGROUP, tid);
    }
#if USE_TDM
    if (wid == 0) {
      if (c + 1 < NCHUNK) __builtin_amdgcn_s_wait_tensorcnt(1); // chunk c complete
      else                __builtin_amdgcn_s_wait_tensorcnt(0);
    }
#endif
    __syncthreads();                  // chunk c visible to all waves

    const int cb = c & 1;
    const int g  = gbase + c;
    const float s = scale[(size_t)orow * NGROUPS + g];
    const float z = zero[(size_t)orow * NGROUPS + g];
    const float bias = -(8388608.0f + z) * s;
    const unsigned char* qr = &qtile[cb][r * QROW_LDS];
    const _Float16* xr = &xh[cb][ln16 * XROW_LDS];

#pragma unroll
    for (int kk = 0; kk < QGROUP; kk += 32) {
      // A fragment (x): halves 0..7 = K kk+8*hl.. , halves 8..15 = K kk+16+8*hl..
      FragA A;
      A.q[0] = *(const u32x4*)(xr + kk + hl * 8);
      A.q[1] = *(const u32x4*)(xr + kk + 16 + hl * 8);
      // B fragment (W col = orow): matching K sub-ranges, 8 nibbles per dword
      const unsigned int b0 = *(const unsigned int*)(qr + (kk >> 1) + hl * 4);
      const unsigned int b1 = *(const unsigned int*)(qr + (kk >> 1) + hl * 4 + 8);
      FragB B;
      dq8(b0, s, bias, &B.u[0]);
      dq8(b1, s, bias, &B.u[4]);
      acc = __builtin_amdgcn_wmma_f32_16x16x32_f16(false, A.v, false, B.v,
                                                   (short)0, acc, false, false);
    }
    __syncthreads();                  // all done reading buffer c before it is refilled
  }

  // D layout: lane(0..15) VGPR j -> M=j, N=ln16. Only M<8 are real batch rows.
  if (hl == 0) {
#pragma unroll
    for (int j = 0; j < 8; ++j)
      __hip_atomic_fetch_add(&out[(size_t)j * OUT_DIM + orow], acc[j],
                             __ATOMIC_RELAXED, __HIP_MEMORY_SCOPE_AGENT);
  }
}

extern "C" void kernel_launch(void* const* d_in, const int* in_sizes, int n_in,
                              void* d_out, int out_size, void* d_ws, size_t ws_size,
                              hipStream_t stream) {
  (void)in_sizes; (void)n_in; (void)out_size; (void)d_ws; (void)ws_size;
  const float* x            = (const float*)d_in[0];
  const unsigned char* qw   = (const unsigned char*)d_in[1];
  const float* scale        = (const float*)d_in[2];
  const float* zero         = (const float*)d_in[3];
  float* out                = (float*)d_out;

  const int n = BATCH * OUT_DIM;
  qlinear_zero_kernel<<<(n + 255) / 256, 256, 0, stream>>>(out, n);

  dim3 grid(OUT_DIM / ROWS_PER_WG, KSPLIT);  // 64 x 4 workgroups
  qlinear_w4a16_wmma_kernel<<<grid, dim3(256), 0, stream>>>(x, qw, scale, zero, out);
}
